// InvariantSlotAttention_77111842832753
// MI455X (gfx1250) — compile-verified, matrix operand order checked by reading, not run on hardware
//
#include <hip/hip_runtime.h>
#include <hip/hip_bf16.h>
#include <math.h>
#include <stdint.h>

// ---------------- model constants ----------------
#define BB 8          // batch
#define SS 8          // slots
#define DD 4096       // spatial positions (64x64)
#define QQ 64         // qdim == hdim
static constexpr float RLOW_C  = 0.1f;               // sqrt(0.01)
static constexpr float RHIGH_C = 0.223606797749979f; // sqrt(0.05)
static constexpr float SMT     = 0.125f;             // 1/sqrt(64)

typedef __attribute__((ext_vector_type(16))) _Float16 v16h;
typedef __attribute__((ext_vector_type(8)))  float    v8f;

__device__ __forceinline__ v8f zero8() { v8f z = {}; return z; }

// ---- CDNA5 async global->LDS staging (ASYNCcnt path) ----
__device__ __forceinline__ void async_ld_b128(unsigned lds_off, const void* g) {
  asm volatile("global_load_async_to_lds_b128 %0, %1, off"
               :: "v"(lds_off), "v"(g) : "memory");
}
__device__ __forceinline__ void wait_async0() {
  asm volatile("s_wait_asynccnt 0" ::: "memory");
}
__device__ __forceinline__ unsigned lds_off_of(const void* p) {
  return (unsigned)(uintptr_t)p;   // LDS aperture: low 32 bits = wave-relative offset
}

// A fragment: 16x32 f16, rows m=0..15 striped over lanes, K within lane regs.
// Layout per CDNA5 ISA 7.12.2 (16-bit A 16x32).
__device__ __forceinline__ v16h frag_a(const _Float16* t, int stride) {
  int lane = threadIdx.x & 31;
  int m  = lane & 15;
  int kh = (lane >> 4) << 3;           // lanes 16-31 hold K+8
  v16h a;
#pragma unroll
  for (int r = 0; r < 8; ++r) {
    int k = ((r & 3) << 1) + ((r >> 2) << 4) + kh; // {0,1},{2,3},{4,5},{6,7},{16..}
    a[2 * r]     = t[m * stride + k];
    a[2 * r + 1] = t[m * stride + k + 1];
  }
  return a;
}

// B fragment: 32x16 f16 [K][N], columns n striped over lanes (mirror of A).
__device__ __forceinline__ v16h frag_b(const _Float16* t, int stride) {
  int lane = threadIdx.x & 31;
  int n  = lane & 15;
  int kh = (lane >> 4) << 3;
  v16h b;
#pragma unroll
  for (int r = 0; r < 8; ++r) {
    int k = ((r & 3) << 1) + ((r >> 2) << 4) + kh;
    b[2 * r]     = t[k * stride + n];
    b[2 * r + 1] = t[(k + 1) * stride + n];
  }
  return b;
}

#define WMMA_F16(A, Bf, C) \
  __builtin_amdgcn_wmma_f32_16x16x32_f16(false, (A), false, (Bf), (short)0, (C), false, false)

// ---------------- generic f32 -> f16 ----------------
__global__ void k_cvt_f16(const float* __restrict__ s, _Float16* __restrict__ d, int n) {
  int i = blockIdx.x * 256 + threadIdx.x;
  if (i < n) d[i] = (_Float16)s[i];
}

// conv weights OIHW f32 -> [tap][ic][oc] f16 (contiguous 8KB per tap)
__global__ void k_wtr(const float* __restrict__ w, _Float16* __restrict__ o) {
  int i = blockIdx.x * 256 + threadIdx.x;  // 25*64*64 = 102400
  if (i >= 25 * 64 * 64) return;
  int tap = i >> 12;
  int r = i & 4095;
  int ic = r >> 6, oc = r & 63;
  o[i] = (_Float16)w[(oc * 64 + ic) * 25 + tap];
}

__global__ void k_zero(float* __restrict__ p, int n) {
  int i = blockIdx.x * 256 + threadIdx.x;
  if (i < n) p[i] = 0.0f;
}

// ---------------- conv0: 1->64 ch, 5x5, pad 2, relu, NHWC f16 out ------------
__global__ __launch_bounds__(256) void k_conv0(const float* __restrict__ x,
                                               const float* __restrict__ w,
                                               const float* __restrict__ bias,
                                               _Float16* __restrict__ out) {
  int idx = blockIdx.x * 256 + threadIdx.x;   // B*4096*64 threads
  int oc  = idx & 63;
  int p   = idx >> 6;
  int pix = p & 4095;
  int b   = p >> 12;
  int py = pix >> 6, px = pix & 63;
  float acc = bias[oc];
#pragma unroll
  for (int dy = 0; dy < 5; ++dy) {
    int iy = py + dy - 2;
    if (iy < 0 || iy > 63) continue;
#pragma unroll
    for (int dx = 0; dx < 5; ++dx) {
      int ix = px + dx - 2;
      if (ix < 0 || ix > 63) continue;
      acc += x[b * 4096 + iy * 64 + ix] * w[oc * 25 + dy * 5 + dx];
    }
  }
  out[idx] = (_Float16)fmaxf(acc, 0.0f);
}

// ---------------- conv1/conv2: implicit GEMM with WMMA ----------------
// M = 32 pixels (one half row), N = 64 out-ch, K = 25 taps * 64 in-ch.
// Weights pre-transposed to [tap][ic][oc] f16: async-staged 8KB blocks.
__global__ __launch_bounds__(128) void k_conv_wmma(const _Float16* __restrict__ in,
                                                   const _Float16* __restrict__ wh,
                                                   const float* __restrict__ bias,
                                                   _Float16* __restrict__ out_h,
                                                   float* __restrict__ out_f32) {
  __shared__ __align__(16) _Float16 at[32][64];   // activations, current tap
  __shared__ __align__(16) _Float16 wt[64][64];   // weights [ic][oc], current tap
  int b    = blockIdx.x >> 7;
  int rest = blockIdx.x & 127;
  int py   = rest >> 1;
  int px0  = (rest & 1) * 32;
  int tid  = threadIdx.x;
  int wv   = tid >> 5;
  int mt   = wv >> 1;
  int nt0  = (wv & 1) * 2;
  unsigned wl = lds_off_of(&wt[0][0]);

  v8f acc0 = zero8(), acc1 = zero8();

  for (int tap = 0; tap < 25; ++tap) {
    int dy = tap / 5 - 2, dx = tap % 5 - 2;
    __syncthreads();
    // async weight tile: contiguous 8KB
    const char* wsrc = (const char*)(wh + tap * 4096);
#pragma unroll
    for (int e = 0; e < 4; ++e) {
      int off = (tid + e * 128) * 16;
      async_ld_b128(wl + off, wsrc + off);
    }
    // activation gather (bounds-checked) via normal VMEM+DS
    for (int e = tid; e < 2048; e += 128) {       // 32 pix x 64 ch
      int m = e >> 6, c = e & 63;
      int iy = py + dy, ix = px0 + m + dx;
      _Float16 v = (_Float16)0.0f;
      if (iy >= 0 && iy < 64 && ix >= 0 && ix < 64)
        v = in[(b * 4096 + iy * 64 + ix) * 64 + c];
      at[m][c] = v;
    }
    wait_async0();
    __syncthreads();
#pragma unroll
    for (int kc = 0; kc < 2; ++kc) {
      v16h a  = frag_a(&at[mt * 16][kc * 32], 64);
      v16h b0 = frag_b(&wt[kc * 32][nt0 * 16], 64);
      acc0 = WMMA_F16(a, b0, acc0);
      v16h b1 = frag_b(&wt[kc * 32][(nt0 + 1) * 16], 64);
      acc1 = WMMA_F16(a, b1, acc1);
    }
  }
  // epilogue: bias + relu
  int lane = tid & 31;
  int n  = lane & 15;
  int mb = (lane >> 4) << 3;
#pragma unroll
  for (int j = 0; j < 8; ++j) {
    int r = mt * 16 + mb + j;            // 0..31
    int base = (b * 4096 + py * 64 + px0 + r) * 64;
    float v0 = fmaxf(acc0[j] + bias[nt0 * 16 + n], 0.0f);
    float v1 = fmaxf(acc1[j] + bias[(nt0 + 1) * 16 + n], 0.0f);
    out_h[base + nt0 * 16 + n]       = (_Float16)v0;
    out_h[base + (nt0 + 1) * 16 + n] = (_Float16)v1;
    if (out_f32) {
      out_f32[base + nt0 * 16 + n]       = v0;
      out_f32[base + (nt0 + 1) * 16 + n] = v1;
    }
  }
}

// ---------------- LayerNorm over 64 (one row / 64-thread block) --------------
__global__ __launch_bounds__(64) void k_ln64(const float* __restrict__ x,
                                             const float* __restrict__ g,
                                             const float* __restrict__ be,
                                             float* __restrict__ y) {
  __shared__ float red[64];
  int row = blockIdx.x, j = threadIdx.x;
  float v = x[row * 64 + j];
  red[j] = v; __syncthreads();
  for (int o = 32; o > 0; o >>= 1) { if (j < o) red[j] += red[j + o]; __syncthreads(); }
  float m = red[0] * (1.0f / 64.0f); __syncthreads();
  red[j] = (v - m) * (v - m); __syncthreads();
  for (int o = 32; o > 0; o >>= 1) { if (j < o) red[j] += red[j + o]; __syncthreads(); }
  float var = red[0] * (1.0f / 64.0f);
  y[row * 64 + j] = (v - m) * rsqrtf(var + 1e-5f) * g[j] + be[j];
}

// ---------------- 64x64 GEMM: out[r,n] = x[r,:]@W + bias (encK / encV) -------
__global__ __launch_bounds__(128) void k_gemm64(const float* __restrict__ x,
                                                const _Float16* __restrict__ wh,
                                                const float* __restrict__ bias,
                                                float* __restrict__ y) {
  __shared__ __align__(16) _Float16 xt[64][64];
  __shared__ __align__(16) _Float16 wt[64][64];
  int row0 = blockIdx.x * 64;
  int tid = threadIdx.x, wv = tid >> 5;
  unsigned wl = lds_off_of(&wt[0][0]);
#pragma unroll
  for (int e = 0; e < 4; ++e) {
    int off = (tid + e * 128) * 16;
    async_ld_b128(wl + off, (const char*)wh + off);
  }
  for (int e = tid; e < 4096; e += 128)
    xt[e >> 6][e & 63] = (_Float16)x[(row0 + (e >> 6)) * 64 + (e & 63)];
  wait_async0();
  __syncthreads();
  v8f acc[4] = {zero8(), zero8(), zero8(), zero8()};
#pragma unroll
  for (int kc = 0; kc < 2; ++kc) {
    v16h a = frag_a(&xt[wv * 16][kc * 32], 64);
#pragma unroll
    for (int nt = 0; nt < 4; ++nt)
      acc[nt] = WMMA_F16(a, frag_b(&wt[kc * 32][nt * 16], 64), acc[nt]);
  }
  int lane = tid & 31, n = lane & 15, mb = (lane >> 4) << 3;
#pragma unroll
  for (int nt = 0; nt < 4; ++nt)
#pragma unroll
    for (int j = 0; j < 8; ++j)
      y[(row0 + wv * 16 + mb + j) * 64 + nt * 16 + n] = acc[nt][j] + bias[nt * 16 + n];
}

// ---------------- slot init: queries, qn, ps (with indexing quirk) -----------
__global__ __launch_bounds__(64) void k_init(const float* __restrict__ mu,
                                             const float* __restrict__ logsig,
                                             const float* __restrict__ noise,
                                             const float* __restrict__ raw,
                                             const float* __restrict__ qg,
                                             const float* __restrict__ qb,
                                             float* __restrict__ queries,
                                             float* __restrict__ qn,
                                             float* __restrict__ ps) {
  __shared__ float red[64];
  int bs = blockIdx.x, s = bs & 7, j = threadIdx.x;
  float q = mu[j] + expf(logsig[j]) * noise[bs * 64 + j];
  queries[bs * 64 + j] = q;
  red[j] = q; __syncthreads();
  for (int o = 32; o > 0; o >>= 1) { if (j < o) red[j] += red[j + o]; __syncthreads(); }
  float m = red[0] * (1.0f / 64.0f); __syncthreads();
  red[j] = (q - m) * (q - m); __syncthreads();
  for (int o = 32; o > 0; o >>= 1) { if (j < o) red[j] += red[j + o]; __syncthreads(); }
  float var = red[0] * (1.0f / 64.0f);
  qn[bs * 64 + j] = (q - m) * rsqrtf(var + 1e-5f) * qg[j] + qb[j];
  if (j < 3) {
    float v = raw[bs * 3 + j];
    if (s < 2) v -= 0.5f;                             // ps.at[:, :2].add(-0.5) (slot axis!)
    if (s == 7) v = (RLOW_C - RHIGH_C) * v + RHIGH_C; // ps.at[:, -1].set(...)
    ps[bs * 3 + j] = v;
  }
}

// ---------------- fused keys -> logits: 128 pixels / 4 waves per block -------
__global__ __launch_bounds__(128) void k_keys_logits(const float* __restrict__ encK,
    const float* __restrict__ qn, const float* __restrict__ ps,
    const float* __restrict__ dw, const float* __restrict__ db,
    const float* __restrict__ qg, const float* __restrict__ qb,
    const _Float16* __restrict__ w1, const float* __restrict__ b1,
    const _Float16* __restrict__ w2, const float* __restrict__ b2,
    float* __restrict__ logits) {
  __shared__ __align__(16) _Float16 kt[128][64];
  __shared__ __align__(16) _Float16 wt1[64][64];
  __shared__ __align__(16) _Float16 wt2[64][64];
  __shared__ __align__(16) _Float16 h1[128][64];
  __shared__ float qns[64];
  int tid  = threadIdx.x;
  int lane = tid & 31;
  int wv   = tid >> 5;
  int dt   = blockIdx.x & 31;    // D/128 = 32 tiles
  int bs   = blockIdx.x >> 5;
  int b    = bs >> 3;
  int d    = dt * 128 + tid;
  // async weight staging: two contiguous 8KB tiles
  unsigned l1 = lds_off_of(&wt1[0][0]);
  unsigned l2 = lds_off_of(&wt2[0][0]);
#pragma unroll
  for (int e = 0; e < 4; ++e) {
    int off = (tid + e * 128) * 16;
    async_ld_b128(l1 + off, (const char*)w1 + off);
    async_ld_b128(l2 + off, (const char*)w2 + off);
  }
  if (tid < 64) qns[tid] = qn[bs * 64 + tid];
  // per-pixel: embed + LN -> kt row
  float px = ps[bs * 3 + 0], py = ps[bs * 3 + 1], sc = ps[bs * 3 + 2];
  float gx = -0.5f + (float)(d & 63) * (1.0f / 63.0f);
  float gy = -0.5f + (float)(d >> 6) * (1.0f / 63.0f);
  float rx = (gx - px) / sc, ry = (gy - py) / sc;
  float rowv[64]; float m = 0.0f;
#pragma unroll
  for (int c = 0; c < 64; ++c) {
    float kv = encK[(b * 4096 + d) * 64 + c] + (rx * dw[c] + ry * dw[64 + c] + db[c]);
    rowv[c] = kv; m += kv;
  }
  m *= (1.0f / 64.0f);
  float var = 0.0f;
#pragma unroll
  for (int c = 0; c < 64; ++c) { float t = rowv[c] - m; var += t * t; }
  float inv = rsqrtf(var * (1.0f / 64.0f) + 1e-5f);
#pragma unroll
  for (int c = 0; c < 64; ++c) kt[tid][c] = (_Float16)((rowv[c] - m) * inv * qg[c] + qb[c]);
  wait_async0();
  __syncthreads();
  int n = lane & 15, mb = (lane >> 4) << 3;
  int rbase = wv * 32;
  // MLP1 (relu) -> h1 (wave-local rows)
#pragma unroll
  for (int mt = 0; mt < 2; ++mt) {
    v8f acc[4] = {zero8(), zero8(), zero8(), zero8()};
#pragma unroll
    for (int kc = 0; kc < 2; ++kc) {
      v16h a = frag_a(&kt[rbase + mt * 16][kc * 32], 64);
#pragma unroll
      for (int nt = 0; nt < 4; ++nt)
        acc[nt] = WMMA_F16(a, frag_b(&wt1[kc * 32][nt * 16], 64), acc[nt]);
    }
#pragma unroll
    for (int nt = 0; nt < 4; ++nt)
#pragma unroll
      for (int j = 0; j < 8; ++j)
        h1[rbase + mt * 16 + mb + j][nt * 16 + n] =
            (_Float16)fmaxf(acc[nt][j] + b1[nt * 16 + n], 0.0f);
  }
  // MLP2 + qn-dot straight from C fragments (no h2 LDS): shuffle reduce
#pragma unroll
  for (int mt = 0; mt < 2; ++mt) {
    v8f acc[4] = {zero8(), zero8(), zero8(), zero8()};
#pragma unroll
    for (int kc = 0; kc < 2; ++kc) {
      v16h a = frag_a(&h1[rbase + mt * 16][kc * 32], 64);
#pragma unroll
      for (int nt = 0; nt < 4; ++nt)
        acc[nt] = WMMA_F16(a, frag_b(&wt2[kc * 32][nt * 16], 64), acc[nt]);
    }
#pragma unroll
    for (int j = 0; j < 8; ++j) {
      float p = 0.0f;
#pragma unroll
      for (int nt = 0; nt < 4; ++nt)
        p += (acc[nt][j] + b2[nt * 16 + n]) * qns[nt * 16 + n];
      p += __shfl_xor(p, 1, 32);
      p += __shfl_xor(p, 2, 32);
      p += __shfl_xor(p, 4, 32);
      p += __shfl_xor(p, 8, 32);   // sum within each 16-lane half (same row group)
      if (n == j) {
        int row = rbase + mt * 16 + mb + j;
        logits[bs * 4096 + dt * 128 + row] = p * SMT;
      }
    }
  }
}

// ---------------- softmax over slot axis (per (b,d)) -------------------------
__global__ __launch_bounds__(256) void k_softmax_slots(const float* __restrict__ logits,
                                                       float* __restrict__ att) {
  int i = blockIdx.x * 256 + threadIdx.x;  // B*D
  int b = i >> 12, d = i & 4095;
  float l[SS], mx = -1e30f;
#pragma unroll
  for (int s = 0; s < SS; ++s) { l[s] = logits[(b * 8 + s) * 4096 + d]; mx = fmaxf(mx, l[s]); }
  float sum = 0.0f;
#pragma unroll
  for (int s = 0; s < SS; ++s) { l[s] = expf(l[s] - mx); sum += l[s]; }
  float r = 1.0f / sum;
#pragma unroll
  for (int s = 0; s < SS; ++s) att[(b * 8 + s) * 4096 + d] = l[s] * r;
}

__device__ __forceinline__ float blockReduce256(float v, float* red) {
  int t = threadIdx.x;
  red[t] = v; __syncthreads();
  for (int o = 128; o > 0; o >>= 1) { if (t < o) red[t] += red[t + o]; __syncthreads(); }
  float r = red[0]; __syncthreads();
  return r;
}

// ---------------- att -> wts, new frames (pos/scale) per (b,s) ---------------
__global__ __launch_bounds__(256) void k_frames(const float* __restrict__ att,
                                                float* __restrict__ wts,
                                                float* __restrict__ ps_new) {
  __shared__ float red[256];
  int bs = blockIdx.x, t = threadIdx.x;
  const float* a = att + bs * 4096;
  float s = 0.0f;
  for (int d = t; d < 4096; d += 256) s += a[d];
  float denom = blockReduce256(s, red);
  float S0 = 0, S1x = 0, S1y = 0, S2 = 0;
  for (int d = t; d < 4096; d += 256) {
    float w = a[d] / denom + 1e-8f;
    wts[bs * 4096 + d] = w;
    float gx = -0.5f + (float)(d & 63) * (1.0f / 63.0f);
    float gy = -0.5f + (float)(d >> 6) * (1.0f / 63.0f);
    S0 += w; S1x += w * gx; S1y += w * gy; S2 += w * (gx * gx + gy * gy);
  }
  S0  = blockReduce256(S0, red);
  S1x = blockReduce256(S1x, red);
  S1y = blockReduce256(S1y, red);
  S2  = blockReduce256(S2, red);
  if (t == 0) {
    float px = S1x, py = S1y;
    float sc2 = S2 - 2.0f * (px * S1x + py * S1y) + (px * px + py * py) * S0;
    ps_new[bs * 3 + 0] = px;
    ps_new[bs * 3 + 1] = py;
    ps_new[bs * 3 + 2] = sqrtf(fmaxf(sc2, 0.0f));
  }
}

// ---------------- fused vals -> upd = sum_d wts*vals (128 px / block) --------
__global__ __launch_bounds__(128) void k_vals_upd(const float* __restrict__ encV,
    const float* __restrict__ wts, const float* __restrict__ ps,
    const float* __restrict__ dw, const float* __restrict__ db,
    const float* __restrict__ qg, const float* __restrict__ qb,
    const _Float16* __restrict__ w1, const float* __restrict__ b1,
    const _Float16* __restrict__ w2, const float* __restrict__ b2,
    float* __restrict__ upd) {
  __shared__ __align__(16) _Float16 kt[128][64];
  __shared__ __align__(16) _Float16 wt1[64][64];
  __shared__ __align__(16) _Float16 wt2[64][64];
  __shared__ __align__(16) _Float16 h1[128][64];
  __shared__ float wsh[128];
  int tid  = threadIdx.x;
  int lane = tid & 31;
  int wv   = tid >> 5;
  int dt   = blockIdx.x & 31;
  int bs   = blockIdx.x >> 5;
  int b    = bs >> 3;
  int d    = dt * 128 + tid;
  unsigned l1 = lds_off_of(&wt1[0][0]);
  unsigned l2 = lds_off_of(&wt2[0][0]);
#pragma unroll
  for (int e = 0; e < 4; ++e) {
    int off = (tid + e * 128) * 16;
    async_ld_b128(l1 + off, (const char*)w1 + off);
    async_ld_b128(l2 + off, (const char*)w2 + off);
  }
  float px = ps[bs * 3 + 0], py = ps[bs * 3 + 1], sc = ps[bs * 3 + 2];
  float gx = -0.5f + (float)(d & 63) * (1.0f / 63.0f);
  float gy = -0.5f + (float)(d >> 6) * (1.0f / 63.0f);
  float rx = (gx - px) / sc, ry = (gy - py) / sc;
  float rowv[64]; float m = 0.0f;
#pragma unroll
  for (int c = 0; c < 64; ++c) {
    float kv = encV[(b * 4096 + d) * 64 + c] + (rx * dw[c] + ry * dw[64 + c] + db[c]);
    rowv[c] = kv; m += kv;
  }
  m *= (1.0f / 64.0f);
  float var = 0.0f;
#pragma unroll
  for (int c = 0; c < 64; ++c) { float t = rowv[c] - m; var += t * t; }
  float inv = rsqrtf(var * (1.0f / 64.0f) + 1e-5f);
#pragma unroll
  for (int c = 0; c < 64; ++c) kt[tid][c] = (_Float16)((rowv[c] - m) * inv * qg[c] + qb[c]);
  wsh[tid] = wts[bs * 4096 + d];
  wait_async0();
  __syncthreads();
  int n = lane & 15, mb = (lane >> 4) << 3;
  int rbase = wv * 32;
#pragma unroll
  for (int mt = 0; mt < 2; ++mt) {
    v8f acc[4] = {zero8(), zero8(), zero8(), zero8()};
#pragma unroll
    for (int kc = 0; kc < 2; ++kc) {
      v16h a = frag_a(&kt[rbase + mt * 16][kc * 32], 64);
#pragma unroll
      for (int nt = 0; nt < 4; ++nt)
        acc[nt] = WMMA_F16(a, frag_b(&wt1[kc * 32][nt * 16], 64), acc[nt]);
    }
#pragma unroll
    for (int nt = 0; nt < 4; ++nt)
#pragma unroll
      for (int j = 0; j < 8; ++j)
        h1[rbase + mt * 16 + mb + j][nt * 16 + n] =
            (_Float16)fmaxf(acc[nt][j] + b1[nt * 16 + n], 0.0f);
  }
  // MLP2 + weighted column sums straight from C fragments
  float colacc[4] = {0.0f, 0.0f, 0.0f, 0.0f};
#pragma unroll
  for (int mt = 0; mt < 2; ++mt) {
    v8f acc[4] = {zero8(), zero8(), zero8(), zero8()};
#pragma unroll
    for (int kc = 0; kc < 2; ++kc) {
      v16h a = frag_a(&h1[rbase + mt * 16][kc * 32], 64);
#pragma unroll
      for (int nt = 0; nt < 4; ++nt)
        acc[nt] = WMMA_F16(a, frag_b(&wt2[kc * 32][nt * 16], 64), acc[nt]);
    }
#pragma unroll
    for (int nt = 0; nt < 4; ++nt)
#pragma unroll
      for (int j = 0; j < 8; ++j)
        colacc[nt] += wsh[rbase + mt * 16 + mb + j] * (acc[nt][j] + b2[nt * 16 + n]);
  }
#pragma unroll
  for (int nt = 0; nt < 4; ++nt) {
    colacc[nt] += __shfl_xor(colacc[nt], 16, 32);   // combine mb=0 and mb=8 halves
    if (lane < 16) atomicAdd(&upd[bs * 64 + nt * 16 + n], colacc[nt]);
  }
}

// ---------------- GRU update + qn recompute ----------------------------------
__global__ __launch_bounds__(64) void k_gru(const float* __restrict__ upd,
    const float* __restrict__ wih, const float* __restrict__ whh,
    const float* __restrict__ bih, const float* __restrict__ bhh,
    const float* __restrict__ qg, const float* __restrict__ qb,
    float* __restrict__ queries, float* __restrict__ qn) {
  __shared__ float xs[64], hs[64], red[64];
  int bs = blockIdx.x, j = threadIdx.x;
  xs[j] = upd[bs * 64 + j];
  hs[j] = queries[bs * 64 + j];
  __syncthreads();
  float xr = bih[j], xz = bih[64 + j], xn = bih[128 + j];
  float hr = bhh[j], hz = bhh[64 + j], hn = bhh[128 + j];
  for (int k = 0; k < 64; ++k) {
    float x = xs[k], h = hs[k];
    xr += x * wih[j * 64 + k];
    xz += x * wih[(64 + j) * 64 + k];
    xn += x * wih[(128 + j) * 64 + k];
    hr += h * whh[j * 64 + k];
    hz += h * whh[(64 + j) * 64 + k];
    hn += h * whh[(128 + j) * 64 + k];
  }
  float r = 1.0f / (1.0f + expf(-(xr + hr)));
  float z = 1.0f / (1.0f + expf(-(xz + hz)));
  float nn = tanhf(xn + r * hn);
  float h = (1.0f - z) * nn + z * hs[j];
  queries[bs * 64 + j] = h;
  red[j] = h; __syncthreads();
  for (int o = 32; o > 0; o >>= 1) { if (j < o) red[j] += red[j + o]; __syncthreads(); }
  float m = red[0] * (1.0f / 64.0f); __syncthreads();
  red[j] = (h - m) * (h - m); __syncthreads();
  for (int o = 32; o > 0; o >>= 1) { if (j < o) red[j] += red[j + o]; __syncthreads(); }
  float var = red[0] * (1.0f / 64.0f);
  qn[bs * 64 + j] = (h - m) * rsqrtf(var + 1e-5f) * qg[j] + qb[j];
}

// ---------------- final head: queries out + slot_feat ------------------------
__global__ __launch_bounds__(64) void k_final(const float* __restrict__ queries,
    const float* __restrict__ f1w, const float* __restrict__ f1b,
    const float* __restrict__ f2w, const float* __restrict__ f2b,
    const float* __restrict__ ps_new,
    float* __restrict__ out_q, float* __restrict__ out_feat) {
  __shared__ float qs[64], hid[64];
  int bs = blockIdx.x, j = threadIdx.x;
  float q = queries[bs * 64 + j];
  out_q[bs * 64 + j] = q;
  qs[j] = q; __syncthreads();
  float h = f1b[j];
  for (int k = 0; k < 64; ++k) h += qs[k] * f1w[k * 64 + j];
  hid[j] = fmaxf(h, 0.0f); __syncthreads();
  if (j < 3) {
    float acc = f2b[j];
    for (int k = 0; k < 64; ++k) acc += hid[k] * f2w[k * 3 + j];
    out_feat[bs * 3 + j] = acc + ps_new[bs * 3 + j];
  }
}

// ================================ host ======================================
extern "C" void kernel_launch(void* const* d_in, const int* in_sizes, int n_in,
                              void* d_out, int out_size, void* d_ws, size_t ws_size,
                              hipStream_t stream) {
  const float* data       = (const float*)d_in[0];
  const float* slot_noise = (const float*)d_in[1];
  const float* ps_raw     = (const float*)d_in[2];
  const float* conv0_w    = (const float*)d_in[3];
  const float* conv0_b    = (const float*)d_in[4];
  const float* conv1_w    = (const float*)d_in[5];
  const float* conv1_b    = (const float*)d_in[6];
  const float* conv2_w    = (const float*)d_in[7];
  const float* conv2_b    = (const float*)d_in[8];
  const float* dataN_g    = (const float*)d_in[9];
  const float* dataN_b    = (const float*)d_in[10];
  const float* queryN_g   = (const float*)d_in[11];
  const float* queryN_b   = (const float*)d_in[12];
  const float* toK_w      = (const float*)d_in[13];
  const float* toK_b      = (const float*)d_in[14];
  const float* toV_w      = (const float*)d_in[15];
  const float* toV_b      = (const float*)d_in[16];
  const float* gru_wih    = (const float*)d_in[17];
  const float* gru_whh    = (const float*)d_in[18];
  const float* gru_bih    = (const float*)d_in[19];
  const float* gru_bhh    = (const float*)d_in[20];
  const float* dense_w    = (const float*)d_in[21];
  const float* dense_b    = (const float*)d_in[22];
  const float* mlp1_w     = (const float*)d_in[23];
  const float* mlp1_b     = (const float*)d_in[24];
  const float* mlp2_w     = (const float*)d_in[25];
  const float* mlp2_b     = (const float*)d_in[26];
  const float* slots_mu   = (const float*)d_in[27];
  const float* slots_ls   = (const float*)d_in[28];
  const float* fin1_w     = (const float*)d_in[29];
  const float* fin1_b     = (const float*)d_in[30];
  const float* fin2_w     = (const float*)d_in[31];
  const float* fin2_b     = (const float*)d_in[32];

  // workspace carve-out
  char* wp = (char*)d_ws;
  auto alloc = [&](size_t bytes) { void* p = wp; wp += (bytes + 255) & ~(size_t)255; return p; };
  const int ACT = BB * DD * QQ;          // 2,097,152
  _Float16* conv1_wh = (_Float16*)alloc(102400 * 2);   // [tap][ic][oc]
  _Float16* conv2_wh = (_Float16*)alloc(102400 * 2);   // [tap][ic][oc]
  _Float16* toK_wh   = (_Float16*)alloc(4096 * 2);
  _Float16* toV_wh   = (_Float16*)alloc(4096 * 2);
  _Float16* mlp1_wh  = (_Float16*)alloc(4096 * 2);
  _Float16* mlp2_wh  = (_Float16*)alloc(4096 * 2);
  _Float16* act0     = (_Float16*)alloc((size_t)ACT * 2);
  _Float16* act1     = (_Float16*)alloc((size_t)ACT * 2);
  float* enc_raw     = (float*)alloc((size_t)ACT * 4);   // reused as encK after LN
  float* enc         = (float*)alloc((size_t)ACT * 4);
  float* encV        = (float*)alloc((size_t)ACT * 4);
  float* logits      = (float*)alloc((size_t)BB * SS * DD * 4);
  float* wts         = (float*)alloc((size_t)BB * SS * DD * 4);
  float* queries     = (float*)alloc(4096 * 4);
  float* qn          = (float*)alloc(4096 * 4);
  float* upd         = (float*)alloc(4096 * 4);
  float* ps_a        = (float*)alloc(192 * 4);
  float* ps_b        = (float*)alloc(192 * 4);
  float* encK        = enc_raw;

  float* out_q    = (float*)d_out;
  float* out_att  = out_q + BB * SS * QQ;       // 4096
  float* out_feat = out_att + BB * SS * DD;     // +262144

  // weight conversions / transposes
  k_wtr<<<400, 256, 0, stream>>>(conv1_w, conv1_wh);
  k_wtr<<<400, 256, 0, stream>>>(conv2_w, conv2_wh);
  k_cvt_f16<<<16, 256, 0, stream>>>(toK_w, toK_wh, 4096);
  k_cvt_f16<<<16, 256, 0, stream>>>(toV_w, toV_wh, 4096);
  k_cvt_f16<<<16, 256, 0, stream>>>(mlp1_w, mlp1_wh, 4096);
  k_cvt_f16<<<16, 256, 0, stream>>>(mlp2_w, mlp2_wh, 4096);

  // encoder
  k_conv0<<<(BB * DD * 64) / 256, 256, 0, stream>>>(data, conv0_w, conv0_b, act0);
  k_conv_wmma<<<BB * 64 * 2, 128, 0, stream>>>(act0, conv1_wh, conv1_b, act1, nullptr);
  k_conv_wmma<<<BB * 64 * 2, 128, 0, stream>>>(act1, conv2_wh, conv2_b, act0, enc_raw);
  k_ln64<<<BB * DD, 64, 0, stream>>>(enc_raw, dataN_g, dataN_b, enc);
  k_gemm64<<<BB * DD / 64, 128, 0, stream>>>(enc, toK_wh, toK_b, encK);
  k_gemm64<<<BB * DD / 64, 128, 0, stream>>>(enc, toV_wh, toV_b, encV);

  // slot init
  k_init<<<BB * SS, 64, 0, stream>>>(slots_mu, slots_ls, slot_noise, ps_raw,
                                     queryN_g, queryN_b, queries, qn, ps_a);

  float* ps_cur = ps_a;
  float* ps_nxt = ps_b;
  const int LBLK = BB * SS * (DD / 128);   // 2048 blocks, 4 waves each

  for (int it = 0; it < 3; ++it) {
    k_keys_logits<<<LBLK, 128, 0, stream>>>(encK, qn, ps_cur, dense_w, dense_b,
                                            queryN_g, queryN_b, mlp1_wh, mlp1_b,
                                            mlp2_wh, mlp2_b, logits);
    k_softmax_slots<<<BB * DD / 256, 256, 0, stream>>>(logits, out_att);
    k_frames<<<BB * SS, 256, 0, stream>>>(out_att, wts, ps_nxt);
    k_zero<<<16, 256, 0, stream>>>(upd, 4096);
    k_vals_upd<<<LBLK, 128, 0, stream>>>(encV, wts, ps_cur, dense_w, dense_b,
                                         queryN_g, queryN_b, mlp1_wh, mlp1_b,
                                         mlp2_wh, mlp2_b, upd);
    k_gru<<<BB * SS, 64, 0, stream>>>(upd, gru_wih, gru_whh, gru_bih, gru_bhh,
                                      queryN_g, queryN_b, queries, qn);
    float* t = ps_cur; ps_cur = ps_nxt; ps_nxt = t;
  }

  // final round: att (to d_out), new_ps, slot_feat, queries
  k_keys_logits<<<LBLK, 128, 0, stream>>>(encK, qn, ps_cur, dense_w, dense_b,
                                          queryN_g, queryN_b, mlp1_wh, mlp1_b,
                                          mlp2_wh, mlp2_b, logits);
  k_softmax_slots<<<BB * DD / 256, 256, 0, stream>>>(logits, out_att);
  k_frames<<<BB * SS, 256, 0, stream>>>(out_att, wts, ps_nxt);
  k_final<<<BB * SS, 64, 0, stream>>>(queries, fin1_w, fin1_b, fin2_w, fin2_b,
                                      ps_nxt, out_q, out_feat);

  (void)in_sizes; (void)n_in; (void)out_size; (void)ws_size;
}
